// DeFormBlock_64321430224901
// MI455X (gfx1250) — matrix operand order repaired and checked by reading, not compiled, not run on hardware
//
#include <hip/hip_runtime.h>
#include <hip/hip_bf16.h>
#include <stdint.h>

typedef __attribute__((ext_vector_type(16))) _Float16 v16h;
typedef __attribute__((ext_vector_type(8)))  float    v8f;

#define B_N   8
#define C_IN  64
#define O_CH  64
#define N_OFF 18               // offset channels (2*K2)
#define N_OFFP 32              // padded to 2 N-tiles
#define HH    128
#define WW    128
#define HW    (HH*WW)          // 16384
#define K2    9
#define KDIM  (C_IN*K2)        // 576
#define KSTEPS (KDIM/32)       // 18
#define APITCH 584             // halves; 584/2=292 dwords; m*292 % 64 distinct for m=0..15
#define OFFPITCH 20            // floats, padded from 18
#define NPIX_CH (B_N*HW)       // 131072 elements per channel

// ---------------- Kernel 1: convert both weight tensors to f16 ----------------
// wh : deform weights (64 x 576)
// who: offset weights padded (32 x 576), rows 18..31 = 0
__global__ void wcvt_k(const float* __restrict__ dw, const float* __restrict__ ow,
                       _Float16* __restrict__ wh, _Float16* __restrict__ who) {
    int i = blockIdx.x * 256 + threadIdx.x;
    if (i < O_CH * KDIM) {
        wh[i] = (_Float16)dw[i];
    } else if (i < (O_CH + N_OFFP) * KDIM) {
        int j = i - O_CH * KDIM;          // row*576 + col, row in [0,32)
        int row = j / KDIM;
        who[j] = (row < N_OFF) ? (_Float16)ow[j] : (_Float16)0.f;
    }
}

// ------- Kernel 2: fused offset-conv + deformable-conv, both via WMMA -------
// One workgroup = 16 pixels (one x-strip) x all output channels. 4 waves.
__global__ __launch_bounds__(128) void deform_fused_k(
        const float* __restrict__ x,
        const _Float16* __restrict__ wh,  const float* __restrict__ dbias,
        const _Float16* __restrict__ who, const float* __restrict__ obias,
        float* __restrict__ out) {
    __shared__ _Float16 A[16 * APITCH];          // A tile (im2col, then sampled)
    __shared__ float    off_lds[16 * OFFPITCH];  // 16 pixels x 18 offset values

    int tile = blockIdx.x;                // ((b*H + y) * (W/16) + xt)
    int xt = tile & 7;
    int y  = (tile >> 3) & (HH - 1);
    int b  = tile >> 10;
    int x0 = xt * 16;
    int t  = threadIdx.x;
    int wave = t >> 5, lane = t & 31;
    int nlo = lane & 15, hi = lane >> 4;

    const float* xb = x + (size_t)b * C_IN * HW;
    const uint32_t* Au = (const uint32_t*)A;
    int arow_h = nlo * APITCH;

    // ---- Phase 1: im2col staging (zero-padded 3x3 taps), f16 ----
    for (int pair = t; pair < 144; pair += 128) {
        int pix = pair / 9, kk = pair - pix * 9;
        int ki = kk / 3, kj = kk - ki * 3;
        int iy = y - 1 + ki;
        int ix = x0 + pix - 1 + kj;
        bool valid = (iy >= 0) & (iy < HH) & (ix >= 0) & (ix < WW);
        int src = iy * WW + ix;
        _Float16* arow = &A[pix * APITCH + kk];
        for (int c = 0; c < C_IN; ++c) {
            float v = valid ? xb[c * HW + src] : 0.f;
            arow[c * 9] = (_Float16)v;
        }
    }
    __syncthreads();

    // ---- Phase 2: offset GEMM on waves 0..1 (N = 32 padded, valid 18) ----
    if (wave < 2) {
        int oc = wave * 16 + nlo;                 // 0..31
        const uint32_t* Whu = (const uint32_t*)who;
        int brow_h = oc * KDIM;
        v8f acc = {};
        for (int ks = 0; ks < KSTEPS; ++ks) {
            int kbase = ks * 32;
            union { v16h v; uint32_t u[8]; } a, bm;
            int ab = arow_h + kbase + (hi ? 16 : 0);
#pragma unroll
            for (int j = 0; j < 8; ++j) {
                a.u[j]     = Au[(ab + 2 * j) >> 1];
            }
            int bb = brow_h + kbase + (hi ? 16 : 0);
#pragma unroll
            for (int j = 0; j < 8; ++j) bm.u[j] = Whu[(bb + 2 * j) >> 1];
            acc = __builtin_amdgcn_wmma_f32_16x16x32_f16(
                      false, a.v, false, bm.v, (short)0, acc, false, false);
        }
        if (oc < N_OFF) {
            float bia = obias[oc];
#pragma unroll
            for (int r = 0; r < 8; ++r) {
                int pix = r + (hi ? 8 : 0);
                off_lds[pix * OFFPITCH + oc] = acc[r] + bia;
            }
        }
    }
    __syncthreads();

    // ---- Phase 3: bilinear sampling with computed offsets, overwrite A ----
    for (int pair = t; pair < 144; pair += 128) {
        int pix = pair / 9, kk = pair - pix * 9;
        int ki = kk / 3, kj = kk - ki * 3;
        int xo = x0 + pix;
        float dy = off_lds[pix * OFFPITCH + 2 * kk];
        float dx = off_lds[pix * OFFPITCH + 2 * kk + 1];
        float sy = (float)(y - 1 + ki) + dy;
        float sx = (float)(xo - 1 + kj) + dx;
        float fy0 = floorf(sy), fx0 = floorf(sx);
        int y0 = (int)fy0, xx0 = (int)fx0;
        int y1 = y0 + 1,   xx1 = xx0 + 1;
        float wy1 = sy - fy0, wy0 = 1.f - wy1;
        float wx1 = sx - fx0, wx0 = 1.f - wx1;
        float m00 = ((y0 >= 0 && y0 < HH && xx0 >= 0 && xx0 < WW) ? 1.f : 0.f) * wy0 * wx0;
        float m01 = ((y0 >= 0 && y0 < HH && xx1 >= 0 && xx1 < WW) ? 1.f : 0.f) * wy0 * wx1;
        float m10 = ((y1 >= 0 && y1 < HH && xx0 >= 0 && xx0 < WW) ? 1.f : 0.f) * wy1 * wx0;
        float m11 = ((y1 >= 0 && y1 < HH && xx1 >= 0 && xx1 < WW) ? 1.f : 0.f) * wy1 * wx1;
        int yc0 = min(max(y0, 0), HH - 1), yc1 = min(max(y1, 0), HH - 1);
        int xc0 = min(max(xx0, 0), WW - 1), xc1 = min(max(xx1, 0), WW - 1);
        int i00 = yc0 * WW + xc0, i01 = yc0 * WW + xc1;
        int i10 = yc1 * WW + xc0, i11 = yc1 * WW + xc1;
        _Float16* arow = &A[pix * APITCH + kk];
        for (int c = 0; c < C_IN; ++c) {
            const float* p = xb + c * HW;
            float v = m00 * p[i00] + m01 * p[i01] + m10 * p[i10] + m11 * p[i11];
            arow[c * 9] = (_Float16)v;
        }
    }
    __syncthreads();

    // ---- Phase 4: deform GEMM, all 4 waves (N-tile per wave) ----
    int oc = wave * 16 + nlo;
    const uint32_t* Whu = (const uint32_t*)wh;
    int brow_h = oc * KDIM;
    v8f acc = {};
    for (int ks = 0; ks < KSTEPS; ++ks) {
        int kbase = ks * 32;
        union { v16h v; uint32_t u[8]; } a, bm;
        int ab = arow_h + kbase + (hi ? 16 : 0);
#pragma unroll
        for (int j = 0; j < 8; ++j) {
            a.u[j]     = Au[(ab + 2 * j) >> 1];
        }
        int bb = brow_h + kbase + (hi ? 16 : 0);
#pragma unroll
        for (int j = 0; j < 8; ++j) bm.u[j] = Whu[(bb + 2 * j) >> 1];
        acc = __builtin_amdgcn_wmma_f32_16x16x32_f16(
                  false, a.v, false, bm.v, (short)0, acc, false, false);
    }

    // D layout: lane -> N=nlo, vgpr r -> M = r + hi*8 ; M = pixel = x - x0
    float bia = dbias[oc];
    size_t obase = (((size_t)b * O_CH + oc) * HH + y) * WW + x0 + (hi ? 8 : 0);
#pragma unroll
    for (int r = 0; r < 8; ++r) out[obase + r] = acc[r] + bia;
}

// ---------------- Kernel 3: per-channel sum / sumsq (one block per channel) ----------------
__global__ void chan_stats_k(const float* __restrict__ y, float* __restrict__ sums) {
    int ch = blockIdx.x;
    float s = 0.f, s2 = 0.f;
    for (int i = threadIdx.x; i < NPIX_CH; i += 256) {
        int b = i >> 14;               // /HW
        int r = i & (HW - 1);
        float v = y[((size_t)(b * O_CH + ch)) * HW + r];
        s += v; s2 += v * v;
    }
    __shared__ float rs[256], rq[256];
    rs[threadIdx.x] = s; rq[threadIdx.x] = s2;
    __syncthreads();
    for (int o = 128; o > 0; o >>= 1) {
        if (threadIdx.x < o) { rs[threadIdx.x] += rs[threadIdx.x + o]; rq[threadIdx.x] += rq[threadIdx.x + o]; }
        __syncthreads();
    }
    if (threadIdx.x == 0) { sums[ch] = rs[0]; sums[O_CH + ch] = rq[0]; }
}

// ---------------- Kernel 4: BN (train) + PReLU, in place ----------------
__global__ void bn_prelu_k(float* __restrict__ y, const float* __restrict__ sums,
                           const float* __restrict__ gamma, const float* __restrict__ beta,
                           const float* __restrict__ pa) {
    int idx = blockIdx.x * 256 + threadIdx.x;
    if (idx >= B_N * O_CH * HW) return;
    int ch = (idx >> 14) & (O_CH - 1);
    const float inv_n = 1.f / (float)NPIX_CH;
    float mean = sums[ch] * inv_n;
    float var  = sums[O_CH + ch] * inv_n - mean * mean;
    float is   = rsqrtf(var + 1e-5f);
    float v = y[idx];
    float r = (v - mean) * is * gamma[ch] + beta[ch];
    y[idx] = (r >= 0.f) ? r : pa[0] * r;
}

extern "C" void kernel_launch(void* const* d_in, const int* in_sizes, int n_in,
                              void* d_out, int out_size, void* d_ws, size_t ws_size,
                              hipStream_t stream) {
    const float* x        = (const float*)d_in[0];
    const float* offset_w = (const float*)d_in[1];
    const float* offset_b = (const float*)d_in[2];
    const float* deform_w = (const float*)d_in[3];
    const float* deform_b = (const float*)d_in[4];
    const float* bn_gamma = (const float*)d_in[5];
    const float* bn_beta  = (const float*)d_in[6];
    const float* prelu_a  = (const float*)d_in[7];
    float* out = (float*)d_out;

    // workspace layout (f16 weights + stats only: ~111 KB)
    char* ws = (char*)d_ws;
    _Float16* wh   = (_Float16*)ws;                                   // 64*576*2 = 73728 B
    _Float16* who  = (_Float16*)(ws + (size_t)O_CH * KDIM * 2);       // 32*576*2 = 36864 B
    float*    sums = (float*)(ws + (size_t)(O_CH + N_OFFP) * KDIM * 2); // 128 f32

    // 1) convert weights to f16 (deform + zero-padded offset weights)
    {
        int n = (O_CH + N_OFFP) * KDIM;   // 55296
        wcvt_k<<<(n + 255) / 256, 256, 0, stream>>>(deform_w, offset_w, wh, who);
    }
    // 2) fused offset conv + deformable conv (WMMA for both GEMMs)
    {
        int tiles = B_N * HH * (WW / 16);   // 8192
        deform_fused_k<<<tiles, 128, 0, stream>>>(x, wh, deform_b, who, offset_b, out);
    }
    // 3) channel stats
    chan_stats_k<<<O_CH, 256, 0, stream>>>(out, sums);
    // 4) BN + PReLU in place
    {
        int n = B_N * O_CH * HW;
        bn_prelu_k<<<(n + 255) / 256, 256, 0, stream>>>(out, sums, bn_gamma, bn_beta, prelu_a);
    }
}
